// ATTNLM_33492154974378
// MI455X (gfx1250) — matrix operand order repaired and checked by reading, not compiled, not run on hardware
//
#include <hip/hip_runtime.h>

// ---------------------------------------------------------------------------
// bf16 WMMA helpers (gfx1250 / CDNA5, wave32)
// ---------------------------------------------------------------------------
typedef __attribute__((ext_vector_type(16))) __bf16 v16bf;
typedef __attribute__((ext_vector_type(8)))  __bf16 v8bf;
typedef __attribute__((ext_vector_type(8)))  float  v8f;

#define DEVI __device__ __forceinline__

DEVI v8f wmma_bf16(v16bf a, v16bf b, v8f c) {
  // D = A(16x32) * B(32x16) + C, f32 accumulate
  return __builtin_amdgcn_wmma_f32_16x16x32_bf16(false, a, false, b,
                                                 (short)0, c, false, false);
}

// A fragment, 16x32 bf16, row-major source with leading dim lda (elements).
// ISA layout: lanes 0-15 -> M=lane, K={0..7,16..23}; lanes 16-31 -> K+=8.
DEVI v16bf load_a_frag(const __bf16* A, int lda, int k_base) {
  int lane = threadIdx.x & 31;
  int grp  = lane >> 4;
  int m    = lane & 15;
  const __bf16* p = A + (size_t)m * lda + k_base + 8 * grp;
  v8bf lo = *(const v8bf*)p;        // K = k_base + 8*grp + [0..7]
  v8bf hi = *(const v8bf*)(p + 16); // K = k_base + 8*grp + 16 + [0..7]
  v16bf a;
#pragma unroll
  for (int i = 0; i < 8; ++i) { a[i] = lo[i]; a[i + 8] = hi[i]; }
  return a;
}

// B fragment, 32x16 bf16, row-major source (ldb elements per K-row).
// ISA layout: lanes 0-15 -> N=lane, K={0..7,16..23}; lanes 16-31 -> K+=8.
DEVI v16bf load_b_frag(const __bf16* B, int ldb, int k_base, int n_base) {
  int lane = threadIdx.x & 31;
  int grp  = lane >> 4;
  const __bf16* p = B + (size_t)(k_base + 8 * grp) * ldb + n_base + (lane & 15);
  v16bf b;
#pragma unroll
  for (int i = 0; i < 8; ++i) {
    b[i]     = p[(size_t)i * ldb];
    b[i + 8] = p[(size_t)(i + 16) * ldb];
  }
  return b;
}

// ---------------------------------------------------------------------------
// Weight transpose + f32->bf16 convert:  out[c*R + r] = bf16(in[r*C + c])
// ---------------------------------------------------------------------------
__global__ void transpose_bf16_kernel(const float* __restrict__ in,
                                      __bf16* __restrict__ out, int R, int C) {
  size_t idx = (size_t)blockIdx.x * 256 + threadIdx.x;
  if (idx >= (size_t)R * C) return;
  int c = (int)(idx / R);
  int r = (int)(idx % R);
  out[idx] = (__bf16)in[(size_t)r * C + c];
}

// ---------------------------------------------------------------------------
// Embedding gather -> bf16 x [16384, 512]
// ---------------------------------------------------------------------------
__global__ void gather_kernel(const int* __restrict__ batch,
                              const float* __restrict__ emb,
                              __bf16* __restrict__ x) {
  int m = blockIdx.x;
  int v = batch[m];
  for (int i = threadIdx.x; i < 512; i += 256)
    x[(size_t)m * 512 + i] = (__bf16)emb[(size_t)v * 512 + i];
}

// ---------------------------------------------------------------------------
// Guard-free bf16 GEMM:  C = A[M,K] * B[K,N] + bias  (f32 or bf16 out)
// block = 256 threads = 8 waves; wave tile = 16 x (16*NSUB)
// block tile = 32 rows x 64*NSUB cols; grid = (N/(64*NSUB), M/32)
// All launches are exactly tile-divisible -> straight-line WMMA inner loop.
// ---------------------------------------------------------------------------
template <int K, int NSUB>
__global__ void __launch_bounds__(256)
gemm_bf16_kernel(const __bf16* __restrict__ A, int lda,
                 const __bf16* __restrict__ B, int ldb,
                 const float* __restrict__ bias,
                 float* __restrict__ outF, __bf16* __restrict__ outB,
                 int ldc) {
  int wave = threadIdx.x >> 5, lane = threadIdx.x & 31;
  int wrow = wave >> 2, wcol = wave & 3;
  int m_tile = (blockIdx.y * 2 + wrow) * 16;
  int n_base = (blockIdx.x * 4 + wcol) * (16 * NSUB);
  const __bf16* Arow = A + (size_t)m_tile * lda;

  v8f acc[NSUB];
#pragma unroll
  for (int s = 0; s < NSUB; ++s)
#pragma unroll
    for (int j = 0; j < 8; ++j) acc[s][j] = 0.f;

#pragma unroll 4
  for (int kb = 0; kb < K; kb += 32) {
    // pull next B k-block toward L2/WGP$ (speculative; stays inside ws)
    __builtin_prefetch((const void*)(B + (size_t)(kb + 32 + lane) * ldb + n_base), 0, 1);
    v16bf a = load_a_frag(Arow, lda, kb);
#pragma unroll
    for (int s = 0; s < NSUB; ++s) {
      v16bf b = load_b_frag(B, ldb, kb, n_base + 16 * s);
      acc[s] = wmma_bf16(a, b, acc[s]);
    }
  }

  int grp = lane >> 4, c0 = lane & 15;
#pragma unroll
  for (int s = 0; s < NSUB; ++s) {
    int col = n_base + 16 * s + c0;
    float bv = bias[col];
#pragma unroll
    for (int j = 0; j < 8; ++j) {
      int row = m_tile + j + 8 * grp;
      float v = acc[s][j] + bv;
      if (outF) outF[(size_t)row * ldc + col] = v;
      else      outB[(size_t)row * ldc + col] = (__bf16)v;
    }
  }
}

// Tail for the logits GEMM: the last 16 columns (N=10000 = 39*256 + 16).
// block = 64 threads = 2 waves; each wave one 16x16 tile; grid = M/32.
__global__ void __launch_bounds__(64)
gemm_tail16_kernel(const __bf16* __restrict__ A, int lda,
                   const __bf16* __restrict__ B, int ldb,
                   const float* __restrict__ bias,
                   float* __restrict__ outF, int ldc, int n0) {
  int wave = threadIdx.x >> 5, lane = threadIdx.x & 31;
  int m_tile = (blockIdx.x * 2 + wave) * 16;
  const __bf16* Arow = A + (size_t)m_tile * lda;
  v8f acc;
#pragma unroll
  for (int j = 0; j < 8; ++j) acc[j] = 0.f;
#pragma unroll 4
  for (int kb = 0; kb < 512; kb += 32) {
    v16bf a = load_a_frag(Arow, lda, kb);
    v16bf b = load_b_frag(B, ldb, kb, n0);
    acc = wmma_bf16(a, b, acc);
  }
  int grp = lane >> 4, c0 = lane & 15;
  int col = n0 + c0;
  float bv = bias[col];
#pragma unroll
  for (int j = 0; j < 8; ++j)
    outF[(size_t)(m_tile + j + 8 * grp) * ldc + col] = acc[j] + bv;
}

// ---------------------------------------------------------------------------
// Persistent RNN scan: h(16x512 bf16, rows 8..15 zero-padded) lives in LDS,
// one workgroup of 32 waves; wave w owns output columns [16w,16w+16).
// h_new = tanh(x_proj + h @ W_hh^T + b_hh); writes z into combined[:,512:]
// and z^T for the attention B-side.
// ---------------------------------------------------------------------------
__global__ void __launch_bounds__(1024)
rnn_kernel(const float* __restrict__ xproj, const __bf16* __restrict__ WhhT,
           const float* __restrict__ bhh, __bf16* __restrict__ combined,
           __bf16* __restrict__ zT) {
  __shared__ __bf16 h[16 * 512];
  int tid = threadIdx.x, wave = tid >> 5, lane = tid & 31;
  for (int i = tid; i < 16 * 512; i += 1024) h[i] = (__bf16)0.f;
  __syncthreads();

  int ncol = wave * 16;
  int col  = ncol + (lane & 15);
  float bb = bhh[col];

  for (int t = 0; t < 2048; ++t) {
    v8f acc;
#pragma unroll
    for (int j = 0; j < 8; ++j) acc[j] = 0.f;
#pragma unroll
    for (int kb = 0; kb < 512; kb += 32) {
      v16bf a = load_a_frag(h, 512, kb);          // ds_load_b128 from LDS
      v16bf b = load_b_frag(WhhT, 512, kb, ncol); // streams from L2 (512KB)
      acc = wmma_bf16(a, b, acc);
    }
    __syncthreads();                              // all reads of h done
    if (lane < 16) {                              // C rows 0..7 = batches
#pragma unroll
      for (int j = 0; j < 8; ++j) {
        float v = tanhf(acc[j] + xproj[((size_t)j * 2048 + t) * 512 + col] + bb);
        __bf16 bv = (__bf16)v;
        h[j * 512 + col] = bv;
        combined[((size_t)j * 2048 + t) * 1024 + 512 + col] = bv;
        zT[((size_t)j * 512 + col) * 2048 + t] = bv;
      }
    }
    __syncthreads();                              // h_new visible
  }
}

// ---------------------------------------------------------------------------
// Flash attention over strict-lower-triangular mask.
// grid = (L/16, B); block = 256 thr = 8 waves. 16-row output tile,
// m processed in 256-wide chunks staged in LDS (S f32, P bf16).
// Wave w accumulates output columns [64w, 64w+64) (4 WMMA C frags).
// Writes att_vec (bf16) into combined[:, :512].
// ---------------------------------------------------------------------------
__global__ void __launch_bounds__(256)
flash_attn_kernel(__bf16* __restrict__ combined, const __bf16* __restrict__ zT) {
  __shared__ float  S[16 * 256];
  __shared__ __bf16 P[16 * 256];
  __shared__ float  red[16 * 16];
  __shared__ float  rowmax[16], rowscale[16], rowsum[16];

  const int L = 2048;
  int b = blockIdx.y;
  int l_base = blockIdx.x * 16;
  int tid = threadIdx.x, wave = tid >> 5, lane = tid & 31;
  int grp = lane >> 4, c0 = lane & 15;
  int row = tid >> 4, sub = tid & 15;
  int gRow = l_base + row;

  const __bf16* zrow = combined + ((size_t)b * L + l_base) * 1024 + 512; // A: z tile rows
  const __bf16* zTb  = zT + (size_t)b * 512 * 2048;                     // B: z^T

  if (tid < 16) { rowmax[tid] = -3.0e38f; rowsum[tid] = 0.f; }
  v8f o[4];
#pragma unroll
  for (int s = 0; s < 4; ++s)
#pragma unroll
    for (int j = 0; j < 8; ++j) o[s][j] = 0.f;
  __syncthreads();

  for (int m_base = 0; m_base < l_base + 15; m_base += 256) {
    // --- scores: 16 tiles of 16x16, 2 per wave, K = 512 ---
#pragma unroll
    for (int tt = 0; tt < 2; ++tt) {
      int nc = (wave * 2 + tt) * 16;
      v8f c;
#pragma unroll
      for (int j = 0; j < 8; ++j) c[j] = 0.f;
#pragma unroll 4
      for (int kb = 0; kb < 512; kb += 32) {
        v16bf a = load_a_frag(zrow, 1024, kb);
        v16bf bf = load_b_frag(zTb, 2048, kb, m_base + nc);
        c = wmma_bf16(a, bf, c);
      }
#pragma unroll
      for (int j = 0; j < 8; ++j) S[(j + 8 * grp) * 256 + nc + c0] = c[j];
    }
    __syncthreads();

    // --- mask + per-row chunk max (16 threads per row, 16 cols each) ---
    float lmax = -3.0e38f;
    int base = row * 256 + sub * 16;
#pragma unroll
    for (int c = 0; c < 16; ++c) {
      int gcol = m_base + sub * 16 + c;
      float s = S[base + c];
      if (gcol >= gRow) { s = -1.0e9f; S[base + c] = s; }
      lmax = fmaxf(lmax, s);
    }
    red[row * 16 + sub] = lmax;
    __syncthreads();
    if (tid < 16) {
      float cm = -3.0e38f;
      for (int i = 0; i < 16; ++i) cm = fmaxf(cm, red[tid * 16 + i]);
      float nm = fmaxf(rowmax[tid], cm);
      rowscale[tid] = __expf(rowmax[tid] - nm);
      rowmax[tid] = nm;
    }
    __syncthreads();

    // --- P = exp(S - newmax), partial row sums ---
    float nm = rowmax[row], psum = 0.f;
#pragma unroll
    for (int c = 0; c < 16; ++c) {
      float p = __expf(S[base + c] - nm);
      P[base + c] = (__bf16)p;
      psum += p;
    }
    red[row * 16 + sub] = psum;
    __syncthreads();
    if (tid < 16) {
      float s16 = 0.f;
      for (int i = 0; i < 16; ++i) s16 += red[tid * 16 + i];
      rowsum[tid] = rowsum[tid] * rowscale[tid] + s16;
    }
    __syncthreads();

    // --- rescale O, then O += P(16x256) @ z_chunk(256x64 per wave) ---
#pragma unroll
    for (int s = 0; s < 4; ++s)
#pragma unroll
      for (int j = 0; j < 8; ++j) o[s][j] *= rowscale[j + 8 * grp];
    const __bf16* zchunk = combined + ((size_t)b * L + m_base) * 1024 + 512;
#pragma unroll 2
    for (int kb = 0; kb < 256; kb += 32) {
      v16bf a = load_a_frag(P, 256, kb);
#pragma unroll
      for (int s = 0; s < 4; ++s) {
        v16bf bf = load_b_frag(zchunk, 1024, kb, wave * 64 + s * 16);
        o[s] = wmma_bf16(a, bf, o[s]);
      }
    }
    __syncthreads();   // protect S/P/rowscale before next chunk
  }

  // --- normalize and store att_vec into combined[:, :512] ---
#pragma unroll
  for (int s = 0; s < 4; ++s)
#pragma unroll
    for (int j = 0; j < 8; ++j) {
      int r = j + 8 * grp;
      int col = wave * 64 + s * 16 + c0;
      combined[((size_t)b * L + l_base + r) * 1024 + col] =
          (__bf16)(o[s][j] / rowsum[r]);
    }
}

// Row l=0: reference softmaxes an all -1e9 row -> uniform over all L positions.
__global__ void fixup_row0_kernel(__bf16* __restrict__ combined) {
  int b = blockIdx.x, h = threadIdx.x;  // 8 blocks x 512 threads
  float s = 0.f;
  for (int t = 0; t < 2048; ++t)
    s += (float)combined[((size_t)b * 2048 + t) * 1024 + 512 + h];
  combined[((size_t)b * 2048) * 1024 + h] = (__bf16)(s * (1.f / 2048.f));
}

// ---------------------------------------------------------------------------
// Host launcher
// ---------------------------------------------------------------------------
extern "C" void kernel_launch(void* const* d_in, const int* in_sizes, int n_in,
                              void* d_out, int out_size, void* d_ws, size_t ws_size,
                              hipStream_t stream) {
  const int*   batch = (const int*)  d_in[0];
  const float* emb   = (const float*)d_in[1];
  const float* W_ih  = (const float*)d_in[2];
  const float* b_ih  = (const float*)d_in[3];
  const float* W_hh  = (const float*)d_in[4];
  const float* b_hh  = (const float*)d_in[5];
  const float* W_c   = (const float*)d_in[6];
  const float* b_c   = (const float*)d_in[7];
  const float* W_d   = (const float*)d_in[8];
  const float* b_d   = (const float*)d_in[9];
  float* out = (float*)d_out;

  char* ws = (char*)d_ws;
  size_t off = 0;
  auto take = [&](size_t bytes) -> void* {
    void* p = (void*)(ws + off);
    off += (bytes + 255) & ~(size_t)255;
    return p;
  };
  __bf16* W_ihT  = (__bf16*)take((size_t)512 * 512 * 2);     // [E,H]
  __bf16* W_hhT  = (__bf16*)take((size_t)512 * 512 * 2);     // [Hin,Hout]
  __bf16* W_cT   = (__bf16*)take((size_t)1024 * 512 * 2);    // [2H,H]
  __bf16* W_dT   = (__bf16*)take((size_t)512 * 10000 * 2);   // [H,V]
  __bf16* x_bf   = (__bf16*)take((size_t)16384 * 512 * 2);   // emb[batch] bf16
  float*  x_proj = (float*) take((size_t)16384 * 512 * 4);   // f32
  __bf16* comb   = (__bf16*)take((size_t)16384 * 1024 * 2);  // [att_vec | z]
  __bf16* zTb    = (__bf16*)take((size_t)8 * 512 * 2048 * 2);// z^T per batch
  __bf16* dec    = (__bf16*)take((size_t)16384 * 512 * 2);   // dec_in bf16

  // 1) weight transposes/converts (B-side contiguous bf16)
  transpose_bf16_kernel<<<(512 * 512 + 255) / 256, 256, 0, stream>>>(W_ih, W_ihT, 512, 512);
  transpose_bf16_kernel<<<(512 * 512 + 255) / 256, 256, 0, stream>>>(W_hh, W_hhT, 512, 512);
  transpose_bf16_kernel<<<(512 * 1024 + 255) / 256, 256, 0, stream>>>(W_c, W_cT, 512, 1024);
  transpose_bf16_kernel<<<(10000 * 512 + 255) / 256, 256, 0, stream>>>(W_d, W_dT, 10000, 512);

  // 2) embedding gather
  gather_kernel<<<16384, 256, 0, stream>>>(batch, emb, x_bf);

  // 3) x_proj = x @ W_ih^T + b_ih   (M=16384, N=512, K=512, f32 out)
  gemm_bf16_kernel<512, 4><<<dim3(2, 512), 256, 0, stream>>>(
      x_bf, 512, W_ihT, 512, b_ih, x_proj, nullptr, 512);

  // 4) serial RNN scan (persistent single workgroup)
  rnn_kernel<<<1, 1024, 0, stream>>>(x_proj, W_hhT, b_hh, comb, zTb);

  // 5) flash attention -> combined[:, :512]
  flash_attn_kernel<<<dim3(128, 8), 256, 0, stream>>>(comb, zTb);
  fixup_row0_kernel<<<8, 512, 0, stream>>>(comb);

  // 6) dec_in = combined @ W_c^T + b_c  (K=1024, bf16 out)
  gemm_bf16_kernel<1024, 4><<<dim3(2, 512), 256, 0, stream>>>(
      comb, 1024, W_cT, 512, b_c, nullptr, dec, 512);

  // 7) logits = dec_in @ W_d^T + b_d  (f32 out -> d_out)
  //    bulk: columns [0, 9984) ; tail: columns [9984, 10000)
  gemm_bf16_kernel<512, 4><<<dim3(39, 512), 256, 0, stream>>>(
      dec, 512, W_dT, 10000, b_d, out, nullptr, 10000);
  gemm_tail16_kernel<<<512, 64, 0, stream>>>(
      dec, 512, W_dT, 10000, b_d, out, 10000, 9984);
}
